// MoRModel_32521492365539
// MI455X (gfx1250) — matrix operand order repaired and checked by reading, not compiled, not run on hardware
//
#include <hip/hip_runtime.h>

// ---------------- model constants ----------------
#define BATCH   2
#define SEQ     2048
#define TOK     4096          // BATCH*SEQ
#define DMODEL  1024
#define NHEADS  16
#define NKV     4
#define DHEAD   64
#define FFI     2816
#define VOCAB   32000

typedef __bf16 bf16_t;
typedef __attribute__((ext_vector_type(16))) __bf16 bf16x16;
typedef __attribute__((ext_vector_type(8)))  __bf16 bf16x8;
typedef __attribute__((ext_vector_type(4)))  __bf16 bf16x4;
typedef __attribute__((ext_vector_type(8)))  float  f32x8;

__device__ inline f32x8 zero8() {
  f32x8 z;
#pragma unroll
  for (int i = 0; i < 8; ++i) z[i] = 0.f;
  return z;
}

__device__ inline bf16x8 pack8(float4 a, float4 b) {
  bf16x8 r;
  r[0] = (bf16_t)a.x; r[1] = (bf16_t)a.y; r[2] = (bf16_t)a.z; r[3] = (bf16_t)a.w;
  r[4] = (bf16_t)b.x; r[5] = (bf16_t)b.y; r[6] = (bf16_t)b.z; r[7] = (bf16_t)b.w;
  return r;
}

__device__ inline bf16x16 pack16(float4 a, float4 b, float4 c, float4 d) {
  bf16x16 r;
  r[0]  = (bf16_t)a.x; r[1]  = (bf16_t)a.y; r[2]  = (bf16_t)a.z; r[3]  = (bf16_t)a.w;
  r[4]  = (bf16_t)b.x; r[5]  = (bf16_t)b.y; r[6]  = (bf16_t)b.z; r[7]  = (bf16_t)b.w;
  r[8]  = (bf16_t)c.x; r[9]  = (bf16_t)c.y; r[10] = (bf16_t)c.z; r[11] = (bf16_t)c.w;
  r[12] = (bf16_t)d.x; r[13] = (bf16_t)d.y; r[14] = (bf16_t)d.z; r[15] = (bf16_t)d.w;
  return r;
}

// ---------------- bf16 WMMA GEMM: C[M,N] = A[M,K] * B[K,N], N/K compile-time ----
// 128x128 block, 8 waves each computing 32x64 (2x4 WMMA tiles).
// LDS tiles pre-swizzled into WMMA fragment layout:
//   A: lane=(row&15)+16*((k>>3)&1), i=(k&7)+(k>=16?8:0)
//   B: lane=(col&15)+16*((k>>3)&1), i=(k&7)+(k>=16?8:0)
#define BM 128
#define BN 128
#define BK 32

template <int N, int K>
__global__ __launch_bounds__(256)
void gemm_bf16(const float* __restrict__ A, const float* __restrict__ B,
               float* __restrict__ C) {
  __shared__ __attribute__((aligned(32))) bf16_t sAf[2][8][32][16];  // 16 KB
  __shared__ __attribute__((aligned(32))) bf16_t sBf[2][8][32][16];  // 16 KB

  const int tid  = threadIdx.x;
  const int lane = tid & 31;
  const int wave = tid >> 5;
  const int half = lane >> 4;
  const int lcol = lane & 15;
  const int wm   = wave & 3;     // 0..3: 32-row group
  const int wn   = wave >> 2;    // 0..1: 64-col group
  const int m0   = blockIdx.y * BM;
  const int n0   = blockIdx.x * BN;

  // A staging: one row, 16 contiguous K per thread -> two 16B LDS stores
  const int ar    = tid >> 1;          // 0..127
  const int ac    = (tid & 1) * 16;    // 0 or 16
  const int amt   = ar >> 4;
  const int alr   = ar & 15;
  const int ai0   = (ac >= 16) ? 8 : 0;
  const float* aptr = A + (size_t)(m0 + ar) * K + ac;

  // B staging: 4 K-rows x 4 N-cols per thread -> four 8B LDS stores
  const int bk  = (tid >> 5) * 4;      // 0,4,...,28
  const int bc  = (tid & 31) * 4;      // 0..124
  const int bnt = bc >> 4;
  const int bls = (bc & 15) + 16 * ((bk >> 3) & 1);
  const int bi0 = (bk & 7) + 8 * (bk >> 4);
  const float* bptr = B + (size_t)bk * N + n0 + bc;

  f32x8 acc[2][4];
#pragma unroll
  for (int i = 0; i < 2; ++i)
#pragma unroll
    for (int j = 0; j < 4; ++j) acc[i][j] = zero8();

  constexpr int nK = K / BK;
  float4 av[4], bv[4];

  // preload + stage K-block 0
#pragma unroll
  for (int p = 0; p < 4; ++p) av[p] = *(const float4*)(aptr + p * 4);
#pragma unroll
  for (int kk = 0; kk < 4; ++kk) bv[kk] = *(const float4*)(bptr + (size_t)kk * N);
  *(bf16x8*)&sAf[0][amt][alr][ai0]      = pack8(av[0], av[1]);
  *(bf16x8*)&sAf[0][amt][alr + 16][ai0] = pack8(av[2], av[3]);
#pragma unroll
  for (int j = 0; j < 4; ++j) {
    bf16x4 pb;
#pragma unroll
    for (int kk = 0; kk < 4; ++kk) pb[kk] = (bf16_t)((const float*)&bv[kk])[j];
    *(bf16x4*)&sBf[0][bnt][bls + j][bi0] = pb;
  }

  for (int kb = 0; kb < nK; ++kb) {
    const int buf = kb & 1;
    __syncthreads();
    if (kb + 1 < nK) {
      const float* ap = aptr + (size_t)(kb + 1) * BK;
#pragma unroll
      for (int p = 0; p < 4; ++p) av[p] = *(const float4*)(ap + p * 4);
      const float* bp = bptr + (size_t)(kb + 1) * BK * N;
#pragma unroll
      for (int kk = 0; kk < 4; ++kk) bv[kk] = *(const float4*)(bp + (size_t)kk * N);
      if (kb + 2 < nK) {
        __builtin_prefetch(aptr + (size_t)(kb + 2) * BK, 0, 0);
        __builtin_prefetch(bptr + (size_t)(kb + 2) * BK * N, 0, 0);
      }
    }
    // compute: single 32B LDS load per WMMA operand, 8 WMMAs per wave
    bf16x16 aF[2], bF[4];
#pragma unroll
    for (int mt = 0; mt < 2; ++mt)
      aF[mt] = *(const bf16x16*)&sAf[buf][wm * 2 + mt][lane][0];
#pragma unroll
    for (int nt = 0; nt < 4; ++nt)
      bF[nt] = *(const bf16x16*)&sBf[buf][wn * 4 + nt][lane][0];
#pragma unroll
    for (int mt = 0; mt < 2; ++mt)
#pragma unroll
      for (int nt = 0; nt < 4; ++nt)
        acc[mt][nt] = __builtin_amdgcn_wmma_f32_16x16x32_bf16(
            false, aF[mt], false, bF[nt], (short)0, acc[mt][nt], false, false);
    // stage next block into the other buffer
    if (kb + 1 < nK) {
      const int nbuf = buf ^ 1;
      *(bf16x8*)&sAf[nbuf][amt][alr][ai0]      = pack8(av[0], av[1]);
      *(bf16x8*)&sAf[nbuf][amt][alr + 16][ai0] = pack8(av[2], av[3]);
#pragma unroll
      for (int j = 0; j < 4; ++j) {
        bf16x4 pb;
#pragma unroll
        for (int kk = 0; kk < 4; ++kk) pb[kk] = (bf16_t)((const float*)&bv[kk])[j];
        *(bf16x4*)&sBf[nbuf][bnt][bls + j][bi0] = pb;
      }
    }
  }

  // epilogue: one base address, all offsets compile-time immediates
  float* Cp = C + (size_t)(m0 + wm * 32 + 8 * half) * N + n0 + wn * 64 + lcol;
#pragma unroll
  for (int mt = 0; mt < 2; ++mt)
#pragma unroll
    for (int nt = 0; nt < 4; ++nt)
#pragma unroll
      for (int r = 0; r < 8; ++r)
        Cp[(size_t)(mt * 16 + r) * N + nt * 16] = acc[mt][nt][r];
}

// ---------------- flash attention: one wave per (b, h, 32-query rows) -----------
// Two 16-row query tiles per wave share every K- and V-fragment (chunk = 32 keys
// = 2 query tiles, so both tiles have identical causal chunk bounds).
// V is pre-transposed to [B, KV, DH, S] so P*V B-fragments are contiguous loads.
__global__ __launch_bounds__(32)
void flash_attn_kernel(const float* __restrict__ Q, const float* __restrict__ Km,
                       const float* __restrict__ Vt, float* __restrict__ O) {
  __shared__ __attribute__((aligned(32))) bf16_t sPf[2][32][16];
  const int lane = threadIdx.x;
  const int half = lane >> 4;
  const int lcol = lane & 15;
  const int qt2  = blockIdx.x;             // 32-row group
  const int h    = blockIdx.y;
  const int bz   = blockIdx.z;
  const int kvh  = h >> 2;                 // GQA: H/KV = 4
  const float qs = 0.125f;                 // 1/sqrt(DH)

  bf16x16 qF[2][2];
#pragma unroll
  for (int mt = 0; mt < 2; ++mt) {
    const float* qp =
        Q + ((size_t)(bz * SEQ + qt2 * 32 + mt * 16 + lcol) * NHEADS + h) * DHEAD;
#pragma unroll
    for (int c = 0; c < 2; ++c) {
      float4 a = *(const float4*)(qp + c * 32 + half * 8);
      float4 b = *(const float4*)(qp + c * 32 + half * 8 + 4);
      float4 cc = *(const float4*)(qp + c * 32 + 16 + half * 8);
      float4 d = *(const float4*)(qp + c * 32 + 16 + half * 8 + 4);
      a.x *= qs; a.y *= qs; a.z *= qs; a.w *= qs;
      b.x *= qs; b.y *= qs; b.z *= qs; b.w *= qs;
      cc.x *= qs; cc.y *= qs; cc.z *= qs; cc.w *= qs;
      d.x *= qs; d.y *= qs; d.z *= qs; d.w *= qs;
      qF[mt][c] = pack16(a, b, cc, d);
    }
  }

  float m_i[2][8], l_i[2][8];
  f32x8 o[2][4];
#pragma unroll
  for (int mt = 0; mt < 2; ++mt) {
#pragma unroll
    for (int r = 0; r < 8; ++r) { m_i[mt][r] = -1e30f; l_i[mt][r] = 0.f; }
#pragma unroll
    for (int t = 0; t < 4; ++t) o[mt][t] = zero8();
  }

  for (int jc = 0; jc <= qt2; ++jc) {
    const int key0 = jc * 32;
    f32x8 sacc[2][2];
#pragma unroll
    for (int mt = 0; mt < 2; ++mt) { sacc[mt][0] = zero8(); sacc[mt][1] = zero8(); }

    // scores = Q * K^T ; each K fragment feeds both query tiles
#pragma unroll
    for (int nt = 0; nt < 2; ++nt) {
      const float* kp =
          Km + ((size_t)(bz * SEQ + key0 + nt * 16 + lcol) * NKV + kvh) * DHEAD;
#pragma unroll
      for (int c = 0; c < 2; ++c) {
        const float* kq = kp + c * 32 + half * 8;
        bf16x16 kF = pack16(*(const float4*)(kq), *(const float4*)(kq + 4),
                            *(const float4*)(kq + 16), *(const float4*)(kq + 20));
#pragma unroll
        for (int mt = 0; mt < 2; ++mt)
          sacc[mt][nt] = __builtin_amdgcn_wmma_f32_16x16x32_bf16(
              false, qF[mt][c], false, kF, (short)0, sacc[mt][nt], false, false);
      }
    }
    // causal mask: only the diagonal chunk is partial
    if (jc == qt2) {
#pragma unroll
      for (int mt = 0; mt < 2; ++mt)
#pragma unroll
        for (int nt = 0; nt < 2; ++nt) {
          int key = key0 + nt * 16 + lcol;
#pragma unroll
          for (int r = 0; r < 8; ++r) {
            int row = qt2 * 32 + mt * 16 + r + 8 * half;
            if (key > row) sacc[mt][nt][r] = -1e30f;
          }
        }
    }
    // online softmax per tile (row reductions across 16-lane half-groups)
#pragma unroll
    for (int mt = 0; mt < 2; ++mt) {
      float scale_r[8];
#pragma unroll
      for (int r = 0; r < 8; ++r) {
        float mx = fmaxf(sacc[mt][0][r], sacc[mt][1][r]);
#pragma unroll
        for (int off = 1; off < 16; off <<= 1)
          mx = fmaxf(mx, __shfl_xor(mx, off, 32));
        float mnew = fmaxf(m_i[mt][r], mx);
        float sc = __expf(m_i[mt][r] - mnew);
        float p0 = __expf(sacc[mt][0][r] - mnew);
        float p1 = __expf(sacc[mt][1][r] - mnew);
        sacc[mt][0][r] = p0; sacc[mt][1][r] = p1;
        float rs = p0 + p1;
#pragma unroll
        for (int off = 1; off < 16; off <<= 1)
          rs += __shfl_xor(rs, off, 32);
        l_i[mt][r] = l_i[mt][r] * sc + rs;
        m_i[mt][r] = mnew;
        scale_r[r] = sc;
      }
#pragma unroll
      for (int t = 0; t < 4; ++t)
#pragma unroll
        for (int r = 0; r < 8; ++r) o[mt][t][r] *= scale_r[r];
    }

    // stage P tiles directly into A-fragment layout
    __syncthreads();
#pragma unroll
    for (int mt = 0; mt < 2; ++mt)
#pragma unroll
      for (int nt = 0; nt < 2; ++nt)
#pragma unroll
        for (int r = 0; r < 8; ++r)
          sPf[mt][(r + 8 * half) + 16 * (lcol >> 3)][(lcol & 7) + 8 * nt] =
              (bf16_t)sacc[mt][nt][r];
    __syncthreads();
    bf16x16 pF[2];
#pragma unroll
    for (int mt = 0; mt < 2; ++mt) pF[mt] = *(const bf16x16*)&sPf[mt][lane][0];

    // ctx += P * V ; each V fragment feeds both query tiles
#pragma unroll
    for (int t = 0; t < 4; ++t) {
      const float* vp = Vt +
          ((size_t)((bz * NKV + kvh) * DHEAD) + t * 16 + lcol) * SEQ +
          key0 + half * 8;
      bf16x16 vF = pack16(*(const float4*)(vp), *(const float4*)(vp + 4),
                          *(const float4*)(vp + 16), *(const float4*)(vp + 20));
#pragma unroll
      for (int mt = 0; mt < 2; ++mt)
        o[mt][t] = __builtin_amdgcn_wmma_f32_16x16x32_bf16(
            false, pF[mt], false, vF, (short)0, o[mt][t], false, false);
    }
  }
#pragma unroll
  for (int mt = 0; mt < 2; ++mt)
#pragma unroll
    for (int t = 0; t < 4; ++t)
#pragma unroll
      for (int r = 0; r < 8; ++r) {
        int row = qt2 * 32 + mt * 16 + r + 8 * half;
        O[((size_t)(bz * SEQ + row) * NHEADS + h) * DHEAD + t * 16 + lcol] =
            o[mt][t][r] / l_i[mt][r];
      }
}

// ---------------- elementwise / reduction kernels ----------------
__global__ __launch_bounds__(256)
void embed_kernel(const int* __restrict__ ids, const float* __restrict__ emb,
                  float* __restrict__ x) {
  int idx = blockIdx.x * 256 + threadIdx.x;        // over TOK*DMODEL/4
  int t = idx >> 8, d4 = idx & 255;
  float4 v = ((const float4*)(emb + (size_t)ids[t] * DMODEL))[d4];
  ((float4*)(x + (size_t)t * DMODEL))[d4] = v;
}

__global__ __launch_bounds__(256)
void rmsnorm_kernel(const float* __restrict__ x, const float* __restrict__ w,
                    float* __restrict__ out) {
  __shared__ float red[256];
  int t = blockIdx.x;
  int d = threadIdx.x * 4;
  const float* xp = x + (size_t)t * DMODEL;
  float4 xv = *(const float4*)(xp + d);
  red[threadIdx.x] = xv.x * xv.x + xv.y * xv.y + xv.z * xv.z + xv.w * xv.w;
  __syncthreads();
  for (int o = 128; o > 0; o >>= 1) {
    if (threadIdx.x < o) red[threadIdx.x] += red[threadIdx.x + o];
    __syncthreads();
  }
  float rs = rsqrtf(red[0] / (float)DMODEL + 1e-6f);
  float4 wv = *(const float4*)(w + d);
  float4 ov;
  ov.x = xv.x * rs * wv.x; ov.y = xv.y * rs * wv.y;
  ov.z = xv.z * rs * wv.z; ov.w = xv.w * rs * wv.w;
  *(float4*)(out + (size_t)t * DMODEL + d) = ov;
}

__global__ __launch_bounds__(256)
void rope_kernel(float* __restrict__ t, int nh) {
  int idx = blockIdx.x * 256 + threadIdx.x;        // over TOK*nh*32
  int d   = idx & 31;
  int hh  = (idx >> 5) % nh;
  int tok = idx / (32 * nh);
  int s   = tok % SEQ;
  float inv = __expf(-(float)d * (9.210340371976184f / 32.f)); // 10000^(-d/32)
  float sn, cs;
  __sincosf((float)s * inv, &sn, &cs);
  float* p = t + (size_t)tok * nh * DHEAD + hh * DHEAD + d;
  float x1 = p[0], x2 = p[32];
  p[0]  = x1 * cs - x2 * sn;
  p[32] = x2 * cs + x1 * sn;
}

// V: [B,S,KV,DH] -> Vt: [B,KV,DH,S]
__global__ __launch_bounds__(256)
void transpose_v_kernel(const float* __restrict__ v, float* __restrict__ vt) {
  int idx = blockIdx.x * 256 + threadIdx.x;   // over BATCH*KV*DH*SEQ (out-linear)
  int s   = idx & (SEQ - 1);
  int d   = (idx >> 11) & (DHEAD - 1);
  int kv  = (idx >> 17) & (NKV - 1);
  int b   = idx >> 19;
  vt[idx] = v[((size_t)(b * SEQ + s) * NKV + kv) * DHEAD + d];
}

__global__ __launch_bounds__(256)
void router_kernel(const float* __restrict__ x, const float* __restrict__ rw,
                   const int* __restrict__ mask, float* __restrict__ scores) {
  __shared__ float red[256];
  int t = blockIdx.x;
  int d = threadIdx.x * 4;
  float4 xv = *(const float4*)(x + (size_t)t * DMODEL + d);
  float4 wv = *(const float4*)(rw + d);
  red[threadIdx.x] = xv.x * wv.x + xv.y * wv.y + xv.z * wv.z + xv.w * wv.w;
  __syncthreads();
  for (int o = 128; o > 0; o >>= 1) {
    if (threadIdx.x < o) red[threadIdx.x] += red[threadIdx.x + o];
    __syncthreads();
  }
  if (threadIdx.x == 0) {
    float s = 1.f / (1.f + __expf(-red[0]));
    scores[t] = mask[t] ? s : -1e30f;
  }
}

// exact top-k via radix select on bit-ordered keys; single workgroup
__global__ __launch_bounds__(1024)
void topk_kernel(const float* __restrict__ scores, int* __restrict__ mask,
                 float* __restrict__ aux, int keep) {
  __shared__ int cnt_s, tie_s;
  __shared__ float sum_s;
  const int tid = threadIdx.x;
  float sc[4]; unsigned key[4];
#pragma unroll
  for (int j = 0; j < 4; ++j) {
    float f = scores[tid * 4 + j];
    sc[j] = f;
    unsigned u = __float_as_uint(f);
    key[j] = (u & 0x80000000u) ? ~u : (u | 0x80000000u);
  }
  unsigned prefix = 0; int need = keep;
  for (int b = 31; b >= 0; --b) {
    unsigned bit = 1u << b;
    unsigned hm = ~(bit - 1u);
    unsigned want = prefix | bit;
    if (tid == 0) cnt_s = 0;
    __syncthreads();
    int c = 0;
#pragma unroll
    for (int j = 0; j < 4; ++j) if ((key[j] & hm) == want) ++c;
    if (c) atomicAdd(&cnt_s, c);
    __syncthreads();
    int cnt = cnt_s;
    if (cnt >= need) prefix = want; else need -= cnt;
    __syncthreads();
  }
  if (tid == 0) { tie_s = 0; sum_s = 0.f; }
  __syncthreads();
#pragma unroll
  for (int j = 0; j < 4; ++j) {
    int idx = tid * 4 + j;
    int m = 0;
    if (key[j] > prefix) { m = 1; atomicAdd(&sum_s, sc[j]); }
    else if (key[j] == prefix) {
      int pos = atomicAdd(&tie_s, 1);
      if (pos < need) { m = 1; atomicAdd(&sum_s, sc[j]); }
    }
    mask[idx] = m;
  }
  __syncthreads();
  if (tid == 0) aux[0] += -(sum_s / (float)keep) * 1e-3f;
}

__global__ __launch_bounds__(256)
void resadd_kernel(float* __restrict__ x, const float* __restrict__ y,
                   const int* __restrict__ mask) {
  int idx = blockIdx.x * 256 + threadIdx.x;        // over TOK*DMODEL/4
  int t = idx >> 8;
  if (mask[t]) {
    float4 a = ((const float4*)x)[idx];
    float4 b = ((const float4*)y)[idx];
    a.x += b.x; a.y += b.y; a.z += b.z; a.w += b.w;
    ((float4*)x)[idx] = a;
  }
}

__global__ __launch_bounds__(256)
void silu_mul_kernel(float* __restrict__ g, const float* __restrict__ u, int n4) {
  int idx = blockIdx.x * 256 + threadIdx.x;
  if (idx < n4) {
    float4 gv = ((const float4*)g)[idx];
    float4 uv = ((const float4*)u)[idx];
    gv.x = gv.x / (1.f + __expf(-gv.x)) * uv.x;
    gv.y = gv.y / (1.f + __expf(-gv.y)) * uv.y;
    gv.z = gv.z / (1.f + __expf(-gv.z)) * uv.z;
    gv.w = gv.w / (1.f + __expf(-gv.w)) * uv.w;
    ((float4*)g)[idx] = gv;
  }
}

__global__ void init_kernel(int* __restrict__ mask, float* __restrict__ aux) {
  int idx = blockIdx.x * 256 + threadIdx.x;
  if (idx < TOK) mask[idx] = 1;
  if (idx == 0) aux[0] = 0.f;
}

__global__ void finish_kernel(const float* __restrict__ aux, float* __restrict__ out_aux) {
  out_aux[0] = aux[0];
}

// ---------------- host orchestration ----------------
extern "C" void kernel_launch(void* const* d_in, const int* in_sizes, int n_in,
                              void* d_out, int out_size, void* d_ws, size_t ws_size,
                              hipStream_t stream) {
  const int*   ids = (const int*)d_in[0];
  const float* emb = (const float*)d_in[1];
  const float* Wq  = (const float*)d_in[2];
  const float* Wk  = (const float*)d_in[3];
  const float* Wv  = (const float*)d_in[4];
  const float* Wo  = (const float*)d_in[5];
  const float* Wg  = (const float*)d_in[6];
  const float* Wu  = (const float*)d_in[7];
  const float* Wd  = (const float*)d_in[8];
  const float* n1  = (const float*)d_in[9];
  const float* n2  = (const float*)d_in[10];
  const float* rw  = (const float*)d_in[11];
  const float* fn  = (const float*)d_in[12];
  const float* lmh = (const float*)d_in[13];

  float* ws = (float*)d_ws;
  size_t off = 0;
  float* x    = ws + off; off += (size_t)TOK * DMODEL;
  float* h    = ws + off; off += (size_t)TOK * DMODEL;
  float* q    = ws + off; off += (size_t)TOK * DMODEL;
  float* kbuf = ws + off; off += (size_t)TOK * NKV * DHEAD;
  float* vbuf = ws + off; off += (size_t)TOK * NKV * DHEAD;
  float* vt   = ws + off; off += (size_t)TOK * NKV * DHEAD;
  float* ctx  = ws + off; off += (size_t)TOK * DMODEL;
  float* g    = ws + off; off += (size_t)TOK * FFI;
  float* u    = ws + off; off += (size_t)TOK * FFI;
  float* scores = ws + off; off += TOK;
  float* aux  = ws + off; off += 16;
  int*   mask = (int*)(ws + off);

  float* logits = (float*)d_out;

  init_kernel<<<TOK / 256, 256, 0, stream>>>(mask, aux);
  embed_kernel<<<(TOK * DMODEL / 4) / 256, 256, 0, stream>>>(ids, emb, x);

  const int keeps[3] = {2744, 1838, 1231};   // int(n*0.67) chain from 4096
  for (int step = 0; step < 3; ++step) {
    router_kernel<<<TOK, 256, 0, stream>>>(x, rw, mask, scores);
    topk_kernel<<<1, 1024, 0, stream>>>(scores, mask, aux, keeps[step]);

    int blk = step % 2;
    const float* Wq_b = Wq + (size_t)blk * DMODEL * (NHEADS * DHEAD);
    const float* Wk_b = Wk + (size_t)blk * DMODEL * (NKV * DHEAD);
    const float* Wv_b = Wv + (size_t)blk * DMODEL * (NKV * DHEAD);
    const float* Wo_b = Wo + (size_t)blk * (NHEADS * DHEAD) * DMODEL;
    const float* Wg_b = Wg + (size_t)blk * DMODEL * FFI;
    const float* Wu_b = Wu + (size_t)blk * DMODEL * FFI;
    const float* Wd_b = Wd + (size_t)blk * FFI * DMODEL;

    rmsnorm_kernel<<<TOK, 256, 0, stream>>>(x, n1 + blk * DMODEL, h);
    gemm_bf16<DMODEL, DMODEL><<<dim3(DMODEL / BN, TOK / BM), 256, 0, stream>>>(h, Wq_b, q);
    gemm_bf16<NKV * DHEAD, DMODEL><<<dim3((NKV * DHEAD) / BN, TOK / BM), 256, 0, stream>>>(h, Wk_b, kbuf);
    gemm_bf16<NKV * DHEAD, DMODEL><<<dim3((NKV * DHEAD) / BN, TOK / BM), 256, 0, stream>>>(h, Wv_b, vbuf);
    rope_kernel<<<(TOK * NHEADS * 32) / 256, 256, 0, stream>>>(q, NHEADS);
    rope_kernel<<<(TOK * NKV * 32) / 256, 256, 0, stream>>>(kbuf, NKV);
    transpose_v_kernel<<<(TOK * NKV * DHEAD) / 256, 256, 0, stream>>>(vbuf, vt);
    flash_attn_kernel<<<dim3(SEQ / 32, NHEADS, BATCH), 32, 0, stream>>>(q, kbuf, vt, ctx);
    gemm_bf16<DMODEL, DMODEL><<<dim3(DMODEL / BN, TOK / BM), 256, 0, stream>>>(ctx, Wo_b, h);
    resadd_kernel<<<(TOK * DMODEL / 4) / 256, 256, 0, stream>>>(x, h, mask);
    rmsnorm_kernel<<<TOK, 256, 0, stream>>>(x, n2 + blk * DMODEL, h);
    gemm_bf16<FFI, DMODEL><<<dim3(FFI / BN, TOK / BM), 256, 0, stream>>>(h, Wg_b, g);
    gemm_bf16<FFI, DMODEL><<<dim3(FFI / BN, TOK / BM), 256, 0, stream>>>(h, Wu_b, u);
    silu_mul_kernel<<<(TOK * FFI / 4) / 256, 256, 0, stream>>>(g, u, TOK * FFI / 4);
    gemm_bf16<DMODEL, FFI><<<dim3(DMODEL / BN, TOK / BM), 256, 0, stream>>>(g, Wd_b, h);
    resadd_kernel<<<(TOK * DMODEL / 4) / 256, 256, 0, stream>>>(x, h, mask);
  }

  rmsnorm_kernel<<<TOK, 256, 0, stream>>>(x, fn, h);
  gemm_bf16<VOCAB, DMODEL><<<dim3(VOCAB / BN, TOK / BM), 256, 0, stream>>>(h, lmh, logits);
  finish_kernel<<<1, 1, 0, stream>>>(aux, logits + (size_t)TOK * VOCAB);
}
